// Encoder_82111184765558
// MI455X (gfx1250) — compile-verified, matrix operand order checked by reading, not run on hardware
//
#include <hip/hip_runtime.h>
#include <hip/hip_bf16.h>

typedef __attribute__((ext_vector_type(16))) __bf16 v16bf;
typedef __attribute__((ext_vector_type(8)))  float  v8f;
typedef __attribute__((ext_vector_type(4)))  int    v4i;

#define UNITS   1024
#define B_      128
#define T_      256
#define FOUR_H  4096
#define NWG     64      // persistent grid: one WG per 16 h-columns
#define NTH     256     // 8 waves (wave32)
#define HSLICE  16
#define KCHUNK  128
#define NCHUNK  (UNITS / KCHUNK)
#define LDS_PITCH 136   // 128 + 8 bf16 pad -> 272B row pitch, bank-conflict free

#if __has_builtin(__builtin_amdgcn_global_load_async_to_lds_b128) && \
    __has_builtin(__builtin_amdgcn_s_wait_asynccnt)
#define HAVE_ASYNC_LDS 1
#else
#define HAVE_ASYNC_LDS 0
#endif

typedef __attribute__((address_space(1))) v4i* as1_v4i;
typedef __attribute__((address_space(3))) v4i* as3_v4i;

union FragU { v16bf v; uint4 q[2]; };

// one-time: U fp32 [1024,4096] row-major -> bf16 column-major [4096][1024]
__global__ void u_convert_kernel(const float* __restrict__ U, __bf16* __restrict__ ucolT) {
  int idx = blockIdx.x * 256 + threadIdx.x;       // 4096*1024 threads
  int k = idx >> 12;                              // 0..1023 (coalesced read)
  int c = idx & 4095;                             // 0..4095
  ucolT[(size_t)c * UNITS + k] = (__bf16)U[(size_t)k * FOUR_H + c];
}

// zero barrier counter + both h ping-pong buffers (h0 = 0)
__global__ void ws_zero_kernel(unsigned* __restrict__ p, int n) {
  int i = blockIdx.x * 256 + threadIdx.x;
  if (i < n) p[i] = 0u;
}

__global__ __launch_bounds__(NTH, 1)
void lstm_persist_kernel(const float* __restrict__ seq,   // [B,T,4]
                         const float* __restrict__ Wm,    // [4,4096]
                         const float* __restrict__ bvec,  // [4096]
                         const __bf16* __restrict__ ucolT,// [4096][1024] col-major bf16
                         __bf16* __restrict__ hbuf0,
                         __bf16* __restrict__ hbuf1,
                         unsigned* __restrict__ counter,
                         float* __restrict__ out) {
  __shared__ __bf16 ldsB[2][64 * LDS_PITCH];       // double-buffered U K-chunks

  const int tid   = threadIdx.x;
  const int wave  = tid >> 5;
  const int lane  = tid & 31;
  const int j     = lane & 15;                     // N column within tile
  const int khalf = (lane >> 4) << 3;              // K offset 0 / 8 per lane half
  const int mhalf = (lane >> 4) << 3;              // C/D row offset 0 / 8
  const int mbase = wave * 16;
  const int n_base = blockIdx.x * HSLICE;

  // staging geometry for this thread (fixed across the whole kernel)
  int st_c[4], st_ko[4];
  const __bf16* st_gp[4];
#pragma unroll
  for (int v = 0; v < 4; ++v) {
    int id = tid + v * NTH;                        // 0..1023 16B groups
    st_c[v]  = id >> 4;                            // local col 0..63
    st_ko[v] = (id & 15) * 8;                      // k offset 0..120
    int gcolc = (st_c[v] >> 4) * UNITS + n_base + (st_c[v] & 15);
    st_gp[v] = ucolT + (size_t)gcolc * UNITS + st_ko[v];
  }

  // hoist x-projection weights: W columns + bias for this lane's gate columns
  float wc[4][4], bias[4];
#pragma unroll
  for (int g = 0; g < 4; ++g) {
    int gcol = g * UNITS + n_base + j;
    bias[g] = bvec[gcol];
#pragma unroll
    for (int f = 0; f < 4; ++f) wc[g][f] = Wm[f * FOUR_H + gcol];
  }

  float cstate[8];
#pragma unroll
  for (int r = 0; r < 8; ++r) cstate[r] = 0.f;

  float* outH = out + (size_t)B_ * T_ * UNITS;
  float* outC = outH + (size_t)B_ * UNITS;

  // stage one U K-chunk into LDS buffer bufIdx (async direct-to-LDS if available)
  auto stage = [&](int kc, int bufIdx) {
#pragma unroll
    for (int v = 0; v < 4; ++v) {
      const __bf16* gp = st_gp[v] + kc;
      __bf16* lp = &ldsB[bufIdx][st_c[v] * LDS_PITCH + st_ko[v]];
#if HAVE_ASYNC_LDS
      __builtin_amdgcn_global_load_async_to_lds_b128(
          (as1_v4i)(void*)gp, (as3_v4i)(void*)lp, 0, 0);
#else
      uint4 val = *(const uint4*)gp;
      *(uint4*)lp = val;
#endif
    }
  };

  for (int t = 0; t < T_; ++t) {
    const __bf16* hprev = (t & 1) ? hbuf1 : hbuf0;
    __bf16*       hnext = (t & 1) ? hbuf0 : hbuf1;

    // accumulators initialized with x-projection + bias (K=4 folded in)
    v8f acc[4];
#pragma unroll
    for (int r = 0; r < 8; ++r) {
      int m = mbase + r + mhalf;
      float4 s = ((const float4*)seq)[m * T_ + t];
#pragma unroll
      for (int g = 0; g < 4; ++g)
        acc[g][r] = bias[g] + s.x * wc[g][0] + s.y * wc[g][1]
                            + s.z * wc[g][2] + s.w * wc[g][3];
    }

    const __bf16* arow = hprev + (size_t)(mbase + j) * UNITS;  // A row = lane&15

    // pipelined K loop: async-stage chunk n+1 while WMMAs consume chunk n
    stage(0, 0);
#if HAVE_ASYNC_LDS
    __builtin_amdgcn_s_wait_asynccnt(0);
#endif
    __syncthreads();

    for (int kci = 0; kci < NCHUNK; ++kci) {
      const int cur = kci & 1;
      if (kci < NCHUNK - 1) stage((kci + 1) * KCHUNK, cur ^ 1);

      const int kc = kci * KCHUNK;
#pragma unroll
      for (int kt = 0; kt < 4; ++kt) {
        int kg = kc + kt * 32 + khalf;
        FragU a;                                  // A: 16x32 bf16 from L2 h buffer
        a.q[0] = *(const uint4*)(arow + kg);
        a.q[1] = *(const uint4*)(arow + kg + 16);
#pragma unroll
        for (int g = 0; g < 4; ++g) {
          FragU bfr;                              // B: 32x16 bf16 from LDS
          const __bf16* bp = &ldsB[cur][(g * 16 + j) * LDS_PITCH + kt * 32 + khalf];
          bfr.q[0] = *(const uint4*)bp;
          bfr.q[1] = *(const uint4*)(bp + 16);
          acc[g] = __builtin_amdgcn_wmma_f32_16x16x32_bf16(
              false, a.v, false, bfr.v, (short)0, acc[g], false, false);
        }
      }

#if HAVE_ASYNC_LDS
      __builtin_amdgcn_s_wait_asynccnt(0);        // next chunk landed in LDS
#endif
      __syncthreads();
    }

    // gate activations + cell update (c lives in registers across all 256 steps)
#pragma unroll
    for (int r = 0; r < 8; ++r) {
      int m  = mbase + r + mhalf;
      int nh = n_base + j;
      float zi = acc[0][r], zf = acc[1][r], zg = acc[2][r], zo = acc[3][r];
      float ig = 1.f / (1.f + __expf(-zi));
      float fg = 1.f / (1.f + __expf(-zf));
      float gg = zg > 0.f ? zg : 0.f;
      float og = 1.f / (1.f + __expf(-zo));
      float cn = fg * cstate[r] + ig * gg;
      cstate[r] = cn;
      float cr = cn > 0.f ? cn : 0.f;
      float h  = og * cr;
      out[((size_t)m * T_ + t) * UNITS + nh] = h;
      hnext[(size_t)m * UNITS + nh] = (__bf16)h;
      if (t == T_ - 1) {
        outH[(size_t)m * UNITS + nh] = h;
        outC[(size_t)m * UNITS + nh] = cn;
      }
    }

    // grid-wide barrier between timesteps (monotone counter, no reset per step)
    if (t < T_ - 1) {
      __threadfence();
      __syncthreads();
      if (tid == 0) {
        atomicAdd(counter, 1u);
        unsigned target = (unsigned)gridDim.x * (unsigned)(t + 1);
        while (__hip_atomic_load(counter, __ATOMIC_RELAXED,
                                 __HIP_MEMORY_SCOPE_AGENT) < target) {
          __builtin_amdgcn_s_sleep(1);
        }
      }
      __syncthreads();
      __builtin_amdgcn_fence(__ATOMIC_ACQUIRE, "agent");
    }
  }
}

extern "C" void kernel_launch(void* const* d_in, const int* in_sizes, int n_in,
                              void* d_out, int out_size, void* d_ws, size_t ws_size,
                              hipStream_t stream) {
  (void)in_sizes; (void)n_in; (void)out_size; (void)ws_size;
  const float* seq = (const float*)d_in[0];
  const float* Wm  = (const float*)d_in[1];
  const float* U   = (const float*)d_in[2];
  const float* b   = (const float*)d_in[3];
  float* out = (float*)d_out;

  char* ws = (char*)d_ws;
  unsigned* counter = (unsigned*)ws;                       // 256 B slot
  __bf16* hbuf0 = (__bf16*)(ws + 256);                     // 256 KB
  __bf16* hbuf1 = (__bf16*)(ws + 256 + 262144);            // 256 KB
  __bf16* ucolT = (__bf16*)(ws + 256 + 2 * 262144);        // 8 MB

  int nz = (256 + 2 * 262144) / 4;
  ws_zero_kernel<<<(nz + 255) / 256, 256, 0, stream>>>((unsigned*)ws, nz);
  u_convert_kernel<<<(4096 * 1024) / 256, 256, 0, stream>>>(U, ucolT);
  lstm_persist_kernel<<<NWG, NTH, 0, stream>>>(seq, Wm, b, ucolT,
                                               hbuf0, hbuf1, counter, out);
}
